// HGCF_2250562863879
// MI455X (gfx1250) — matrix operand order repaired and checked by reading, not compiled, not run on hardware
//
#include <hip/hip_runtime.h>
#include <hip/hip_bf16.h>

#define N_NODES_C 100000
#define NU_C      60000
#define D_C       64
#define E_C       8
#define EPSF      1e-07f
#define MAXN      10.0f
#define MARGIN_C  0.1f
#define BETA_C    0.1f

typedef __attribute__((ext_vector_type(16))) __bf16 bf16x16;
typedef __attribute__((ext_vector_type(8)))  float  f32x8;

// ---------------------------------------------------------------- helpers
// Build all A-fragments (16x32 bf16 each) for a 16xK f32 LDS tile ONCE.
// Layout: lane<16: K = kb+0..15; lane>=16: K = kb+16..31 (contiguous, matches B).
// Native __bf16 cast -> hardware cvt ops.
__device__ __forceinline__ void build_afrags(const float* tile, int ld, int lane,
                                             int K, bf16x16* af) {
    int m = lane & 15, h = (lane >> 4) & 1;
#pragma unroll
    for (int kb = 0; kb < K; kb += 32) {
        const float* p = tile + m * ld + kb + h * 16;
        bf16x16 a;
#pragma unroll
        for (int i = 0; i < 8; ++i) a[i] = (__bf16)p[i];
#pragma unroll
        for (int i = 0; i < 8; ++i) a[8 + i] = (__bf16)p[8 + i];
        af[kb >> 5] = a;
    }
}

// 16x16 f32 D-tile using prebuilt A-fragments; W is bf16 [N][K] row-major.
__device__ __forceinline__ f32x8 gemm_frag(const bf16x16* af, const __bf16* w, int ldk,
                                           int lane, int nb, int K) {
    f32x8 c = {0.f, 0.f, 0.f, 0.f, 0.f, 0.f, 0.f, 0.f};
    int n = lane & 15, h = (lane >> 4) & 1;
#pragma unroll
    for (int kb = 0; kb < K; kb += 32) {
        bf16x16 b = *(const bf16x16*)(w + (size_t)(nb + n) * ldk + kb + h * 16);
        c = __builtin_amdgcn_wmma_f32_16x16x32_bf16(false, af[kb >> 5], false, b,
                                                    (short)0, c, false, false);
    }
    return c;
}

// ---------------------------------------------------------------- weight pre-pack
// bf16 layout (elements): [0,32768) tW [e][o][d] | [32768,36864) muW [o][d]
// [36864,40960) lvW | [40960,49152) f1W [128][64] | [49152,57344) f2W [64][128]
// [57344,90112) dW [e][o][d] | [90112,91136) tgW pad [16][64] | [91136,92160) dgW pad
__global__ void convert_weights(const float* tW, const float* muW, const float* lvW,
                                const float* f1W, const float* f2W, const float* dW,
                                const float* tgW, const float* dgW,
                                unsigned short* wb) {
    int i = blockIdx.x * blockDim.x + threadIdx.x;
    float v = 0.f;
    bool valid = true;
    if (i < 32768) {
        int e = i >> 12, rem = i & 4095, o = rem >> 6, d = rem & 63;
        v = tW[e * 4096 + d * 64 + o];
    } else if (i < 36864) {
        int j = i - 32768, o = j >> 6, d = j & 63;
        v = muW[d * 64 + o];
    } else if (i < 40960) {
        int j = i - 36864, o = j >> 6, d = j & 63;
        v = lvW[d * 64 + o];
    } else if (i < 49152) {
        int j = i - 40960, o = j >> 6, d = j & 63;       // o:0..127
        v = f1W[d * 128 + o];
    } else if (i < 57344) {
        int j = i - 49152, o = j >> 7, d = j & 127;      // [64][128]
        v = f2W[d * 64 + o];
    } else if (i < 90112) {
        int j = i - 57344, e = j >> 12, rem = j & 4095, o = rem >> 6, d = rem & 63;
        v = dW[e * 4096 + d * 64 + o];
    } else if (i < 91136) {
        int j = i - 90112, o = j >> 6, d = j & 63;       // o:0..15, pad o>=8 with 0
        v = (o < E_C) ? tgW[d * E_C + o] : 0.f;
    } else if (i < 92160) {
        int j = i - 91136, o = j >> 6, d = j & 63;
        v = (o < E_C) ? dgW[d * E_C + o] : 0.f;
    } else {
        valid = false;
    }
    if (valid) {
        __bf16 b = (__bf16)v;
        unsigned short u; __builtin_memcpy(&u, &b, 2);
        wb[i] = u;
    }
}

// ---------------------------------------------------------------- fused VAE/MoE kernel
// One wave (32 threads) per 16-node tile. All GEMMs (incl. gate projections) via
// v_wmma_f32_16x16x32_bf16, tile stays resident in LDS end-to-end.
__global__ void __launch_bounds__(32)
hgcf_dense(const float* __restrict__ emb_user, const float* __restrict__ emb_item,
           const float* __restrict__ eps,
           const float* __restrict__ tb,  const float* __restrict__ tgb,
           const float* __restrict__ mub, const float* __restrict__ lvb,
           const float* __restrict__ f1b, const float* __restrict__ f2b,
           const float* __restrict__ db,  const float* __restrict__ dgb,
           const unsigned short* __restrict__ wb_u16,
           float* __restrict__ m_out, float* recon_sum, float* kl_sum) {
    const __bf16* wb    = (const __bf16*)wb_u16;
    const __bf16* tWbf  = wb;
    const __bf16* muWbf = wb + 32768;
    const __bf16* lvWbf = wb + 36864;
    const __bf16* f1Wbf = wb + 40960;
    const __bf16* f2Wbf = wb + 49152;
    const __bf16* dWbf  = wb + 57344;
    const __bf16* tgWbf = wb + 90112;
    const __bf16* dgWbf = wb + 91136;

    __shared__ float A[16][72];     // t -> mu -> u -> h -> h2
    __shared__ float Bt[16][72];    // m (kept until recon target)
    __shared__ float Cs[16][136];   // logvar -> h1(128) -> decoded
    __shared__ float G[16][8];      // gate weights
    __shared__ float red[2];        // recon, kl partials

    int lane = threadIdx.x;
    int row0 = blockIdx.x * 16;
    int n = lane & 15, mb = ((lane >> 4) & 1) * 8;

    // ---- stage 1: t = logmap0(emb)
    if (lane < 2) red[lane] = 0.f;
    if (lane < 16) {
        int g = row0 + lane;
        const float* src = (g < NU_C) ? emb_user + (size_t)g * D_C
                                      : emb_item + (size_t)(g - NU_C) * D_C;
        float x0 = src[0];
        float alpha = fmaxf(x0, 1.0f + EPSF);
        float coef = acoshf(alpha) / sqrtf(fmaxf(alpha * alpha - 1.0f, EPSF));
        A[lane][0] = 0.f;
        for (int j = 1; j < D_C; ++j) A[lane][j] = coef * src[j];
    }
    __syncthreads();

    // ---- stage 2: encoder gates via WMMA (zero-padded 64x16 gate matrix)
    bf16x16 afr[4];                       // reused: t, m, h, h1(h2 source)
    build_afrags(&A[0][0], 72, lane, 64, afr);
    {
        f32x8 c = gemm_frag(afr, tgWbf, 64, lane, 0, 64);
        if (n < E_C) {
#pragma unroll
            for (int i = 0; i < 8; ++i) G[mb + i][n] = c[i] + tgb[n];
        }
    }
    __syncthreads();
    if (lane < 16) {
        float mx = -1e30f;
        for (int e = 0; e < E_C; ++e) mx = fmaxf(mx, G[lane][e]);
        float s = 0.f, ex[E_C];
        for (int e = 0; e < E_C; ++e) { ex[e] = expf(G[lane][e] - mx); s += ex[e]; }
        for (int e = 0; e < E_C; ++e) G[lane][e] = ex[e] / s;
    }
    __syncthreads();

    // ---- stage 3: m = MoE_encode(t)  (A-frags reused across 8 experts x 4 n-tiles)
    for (int idx = lane; idx < 1024; idx += 32) Bt[idx >> 6][idx & 63] = 0.f;
    __syncthreads();
    for (int e = 0; e < E_C; ++e) {
        const __bf16* we = tWbf + e * 4096;
        const float* tbv = tb + e * 64;
        for (int nb = 0; nb < 64; nb += 16) {
            f32x8 c = gemm_frag(afr, we, 64, lane, nb, 64);
#pragma unroll
            for (int i = 0; i < 8; ++i) {
                int rr = mb + i;
                Bt[rr][nb + n] += G[rr][e] * (c[i] + tbv[nb + n]);
            }
        }
    }
    __syncthreads();
    for (int idx = lane; idx < 1024; idx += 32) {
        int rr = idx >> 6, jj = idx & 63;
        m_out[(size_t)(row0 + rr) * D_C + jj] = Bt[rr][jj];
    }

    // ---- stage 4: mu = m@muW+mub (->A), logvar = m@lvW+lvb (->Cs)
    build_afrags(&Bt[0][0], 72, lane, 64, afr);   // m-fragments, reused for mu+lv
    for (int nb = 0; nb < 64; nb += 16) {
        f32x8 c = gemm_frag(afr, muWbf, 64, lane, nb, 64);
#pragma unroll
        for (int i = 0; i < 8; ++i) A[mb + i][nb + n] = c[i] + mub[nb + n];
    }
    for (int nb = 0; nb < 64; nb += 16) {
        f32x8 c = gemm_frag(afr, lvWbf, 64, lane, nb, 64);
#pragma unroll
        for (int i = 0; i < 8; ++i) Cs[mb + i][nb + n] = c[i] + lvb[nb + n];
    }
    __syncthreads();

    // ---- stage 5: KL partial + u = mu + eps*std  (->A)
    {
        float klp = 0.f;
        for (int idx = lane; idx < 1024; idx += 32) {
            int rr = idx >> 6, jj = idx & 63;
            float muv = A[rr][jj], lvv = Cs[rr][jj];
            klp += 1.0f + lvv - muv * muv - expf(lvv);
            A[rr][jj] = muv + eps[(size_t)(row0 + rr) * D_C + jj] * expf(0.5f * lvv);
        }
        atomicAdd(&red[1], klp);
    }
    __syncthreads();

    // ---- stage 6: h = logmap0(expmap0(u, project))  (->A, per row)
    if (lane < 16) {
        float nn = 0.f;
        for (int j = 1; j < D_C; ++j) { float v = A[lane][j]; nn += v * v; }
        float nv = fmaxf(sqrtf(nn), EPSF);
        float s = sinhf(nv) / nv;
        float f = s * fminf(1.0f, MAXN / fmaxf(s * nv, EPSF));
        float fn = f * nv;
        float x0 = sqrtf(1.0f + fn * fn);
        float alpha = fmaxf(x0, 1.0f + EPSF);
        float coef = acoshf(alpha) / sqrtf(fmaxf(alpha * alpha - 1.0f, EPSF));
        float cf = coef * f;
        A[lane][0] = 0.f;
        for (int j = 1; j < D_C; ++j) A[lane][j] *= cf;
    }
    __syncthreads();

    // ---- stage 7: h1 = relu(h @ f1W + f1b)  (16x128 -> Cs)
    build_afrags(&A[0][0], 72, lane, 64, afr);    // h-fragments, reused for 8 n-tiles
    for (int nb = 0; nb < 128; nb += 16) {
        f32x8 c = gemm_frag(afr, f1Wbf, 64, lane, nb, 64);
#pragma unroll
        for (int i = 0; i < 8; ++i) Cs[mb + i][nb + n] = fmaxf(c[i] + f1b[nb + n], 0.f);
    }
    __syncthreads();

    // ---- stage 8: h2 = h1 @ f2W + f2b  (K=128 -> A)
    build_afrags(&Cs[0][0], 136, lane, 128, afr); // h1-fragments (4 kb), 4 n-tiles
    for (int nb = 0; nb < 64; nb += 16) {
        f32x8 c = gemm_frag(afr, f2Wbf, 128, lane, nb, 128);
#pragma unroll
        for (int i = 0; i < 8; ++i) A[mb + i][nb + n] = c[i] + f2b[nb + n];
    }
    __syncthreads();

    // ---- stage 9: decoder gates via WMMA
    build_afrags(&A[0][0], 72, lane, 64, afr);    // h2-fragments, reused in stage 10
    {
        f32x8 c = gemm_frag(afr, dgWbf, 64, lane, 0, 64);
        if (n < E_C) {
#pragma unroll
            for (int i = 0; i < 8; ++i) G[mb + i][n] = c[i] + dgb[n];
        }
    }
    __syncthreads();
    if (lane < 16) {
        float mx = -1e30f;
        for (int e = 0; e < E_C; ++e) mx = fmaxf(mx, G[lane][e]);
        float s = 0.f, ex[E_C];
        for (int e = 0; e < E_C; ++e) { ex[e] = expf(G[lane][e] - mx); s += ex[e]; }
        for (int e = 0; e < E_C; ++e) G[lane][e] = ex[e] / s;
    }
    __syncthreads();

    // ---- stage 10: hd = MoE_decode(h2) (-> Cs[:,0:64])
    for (int idx = lane; idx < 1024; idx += 32) Cs[idx >> 6][idx & 63] = 0.f;
    __syncthreads();
    for (int e = 0; e < E_C; ++e) {
        const __bf16* we = dWbf + e * 4096;
        const float* dbv = db + e * 64;
        for (int nb = 0; nb < 64; nb += 16) {
            f32x8 c = gemm_frag(afr, we, 64, lane, nb, 64);
#pragma unroll
            for (int i = 0; i < 8; ++i) {
                int rr = mb + i;
                Cs[rr][nb + n] += G[rr][e] * (c[i] + dbv[nb + n]);
            }
        }
    }
    __syncthreads();

    // ---- stage 11: recon loss partial: sqdist(expmap0(hd,proj), expmap0(m))
    if (lane < 16) {
        float nn = 0.f;
        for (int j = 1; j < D_C; ++j) { float v = Cs[lane][j]; nn += v * v; }
        float nv = fmaxf(sqrtf(nn), EPSF);
        float s = sinhf(nv) / nv;
        float f = s * fminf(1.0f, MAXN / fmaxf(s * nv, EPSF));
        float fn = f * nv;
        float x0r = sqrtf(1.0f + fn * fn);
        float nm2 = 0.f;
        for (int j = 1; j < D_C; ++j) { float v = Bt[lane][j]; nm2 += v * v; }
        float nm = fmaxf(sqrtf(nm2), EPSF);
        float st = sinhf(nm) / nm;
        float x0t = coshf(nm);
        float dt = 0.f;
        for (int j = 1; j < D_C; ++j) dt += Cs[lane][j] * Bt[lane][j];
        float inner = -x0r * x0t + f * st * dt;
        float d = acoshf(fmaxf(-inner, 1.0f + EPSF));
        atomicAdd(&red[0], d * d);
    }
    __syncthreads();
    if (lane == 0) {
        atomicAdd(recon_sum, red[0]);
        atomicAdd(kl_sum, red[1]);
    }
}

// ---------------------------------------------------------------- SpMM / misc
__global__ void zero_f32(float* p, int n) {
    int i = blockIdx.x * blockDim.x + threadIdx.x;
    if (i < n) p[i] = 0.f;
}

__global__ void accum_add(float* dst, const float* src, int n) {
    int i = blockIdx.x * blockDim.x + threadIdx.x;
    if (i < n) dst[i] += src[i];
}

// dst[rows[i]] += vals[i] * src[cols[i]] ; 4 threads per nnz, 16 dims each.
__global__ void spmm_scatter(const int* __restrict__ rows, const int* __restrict__ cols,
                             const float* __restrict__ vals, const float* __restrict__ src,
                             float* __restrict__ dst, int nnz) {
    long idx = (long)blockIdx.x * blockDim.x + threadIdx.x;
    int i = (int)(idx >> 2), q = (int)(idx & 3);
    if (i >= nnz) return;
    if (i + 4096 < nnz) {   // stream-ahead prefetch (global_prefetch_b8)
        __builtin_prefetch(&rows[i + 4096], 0, 1);
        __builtin_prefetch(&cols[i + 4096], 0, 1);
    }
    int r = rows[i], c = cols[i];
    float v = vals[i];
    const float4* s = (const float4*)(src + (size_t)c * 64 + q * 16);
    float4 a0 = s[0], a1 = s[1], a2 = s[2], a3 = s[3];
    float* d = dst + (size_t)r * 64 + q * 16;
    atomicAdd(d + 0,  v * a0.x); atomicAdd(d + 1,  v * a0.y);
    atomicAdd(d + 2,  v * a0.z); atomicAdd(d + 3,  v * a0.w);
    atomicAdd(d + 4,  v * a1.x); atomicAdd(d + 5,  v * a1.y);
    atomicAdd(d + 6,  v * a1.z); atomicAdd(d + 7,  v * a1.w);
    atomicAdd(d + 8,  v * a2.x); atomicAdd(d + 9,  v * a2.y);
    atomicAdd(d + 10, v * a2.z); atomicAdd(d + 11, v * a2.w);
    atomicAdd(d + 12, v * a3.x); atomicAdd(d + 13, v * a3.y);
    atomicAdd(d + 14, v * a3.z); atomicAdd(d + 15, v * a3.w);
}

__global__ void expmap_rows(const float* __restrict__ src, float* __restrict__ dst, int n) {
    int r = blockIdx.x * blockDim.x + threadIdx.x;
    if (r >= n) return;
    const float* u = src + (size_t)r * 64;
    float* o = dst + (size_t)r * 64;
    float nn = 0.f;
    for (int j = 1; j < 64; ++j) { float v = u[j]; nn += v * v; }
    float nv = fmaxf(sqrtf(nn), EPSF);
    float s = sinhf(nv) / nv;
    float f = s * fminf(1.0f, MAXN / fmaxf(s * nv, EPSF));
    float ss = 0.f;
    for (int j = 1; j < 64; ++j) { float v = f * u[j]; o[j] = v; ss += v * v; }
    o[0] = sqrtf(1.0f + ss);
}

__device__ __forceinline__ float hyp_sqdist(const float* x, const float* y) {
    float dot = 0.f;
    const float4* xa = (const float4*)x;
    const float4* ya = (const float4*)y;
#pragma unroll
    for (int q = 0; q < 16; ++q) {
        float4 u = xa[q], v = ya[q];
        dot += u.x * v.x + u.y * v.y + u.z * v.z + u.w * v.w;
    }
    float inner = dot - 2.0f * x[0] * y[0];   // -x0*y0 + <spatial>
    float d = acoshf(fmaxf(-inner, 1.0f + EPSF));
    return d * d;
}

__global__ void margin_kernel(const int* __restrict__ tr, const float* __restrict__ H,
                              float* margin_sum, int T) {
    int i = blockIdx.x * blockDim.x + threadIdx.x;
    float l = 0.f;
    if (i < T) {
        int a = tr[3 * i], p = tr[3 * i + 1], n = tr[3 * i + 2];
        const float* xa = H + (size_t)a * 64;
        float dap = hyp_sqdist(xa, H + (size_t)p * 64);
        float dan = hyp_sqdist(xa, H + (size_t)n * 64);
        float m = dap - dan + MARGIN_C;
        l = m > 0.f ? m : 0.f;
    }
    __shared__ float sred[256];
    sred[threadIdx.x] = l;
    __syncthreads();
    for (int s = 128; s > 0; s >>= 1) {
        if (threadIdx.x < s) sred[threadIdx.x] += sred[threadIdx.x + s];
        __syncthreads();
    }
    if (threadIdx.x == 0) atomicAdd(margin_sum, sred[0]);
}

__global__ void finalize(const float* scal, float* out) {
    float recon = scal[0] / (float)N_NODES_C;
    float kl = -0.5f * scal[1] / (float)(N_NODES_C * 64);
    out[0] = scal[2] + BETA_C * (recon + kl);
}

// ---------------------------------------------------------------- launch
extern "C" void kernel_launch(void* const* d_in, const int* in_sizes, int n_in,
                              void* d_out, int out_size, void* d_ws, size_t ws_size,
                              hipStream_t stream) {
    const float* emb_user = (const float*)d_in[0];
    const float* emb_item = (const float*)d_in[1];
    const int*   adj_rows = (const int*)d_in[2];
    const int*   adj_cols = (const int*)d_in[3];
    const float* adj_vals = (const float*)d_in[4];
    const int*   triples  = (const int*)d_in[5];
    const float* eps      = (const float*)d_in[6];
    const float* tW  = (const float*)d_in[7];
    const float* tb  = (const float*)d_in[8];
    const float* tgW = (const float*)d_in[9];
    const float* tgb = (const float*)d_in[10];
    const float* muW = (const float*)d_in[11];
    const float* mub = (const float*)d_in[12];
    const float* lvW = (const float*)d_in[13];
    const float* lvb = (const float*)d_in[14];
    const float* f1W = (const float*)d_in[15];
    const float* f1b = (const float*)d_in[16];
    const float* f2W = (const float*)d_in[17];
    const float* f2b = (const float*)d_in[18];
    const float* dW  = (const float*)d_in[19];
    const float* db  = (const float*)d_in[20];
    const float* dgW = (const float*)d_in[21];
    const float* dgb = (const float*)d_in[22];

    char* ws = (char*)d_ws;
    unsigned short* wb = (unsigned short*)(ws + 0);       // 92160 bf16 = 180 KB
    float* scal = (float*)(ws + 196608);                  // [recon, kl_raw, margin]
    const size_t NB = (size_t)N_NODES_C * 64 * sizeof(float);
    float* mbuf = (float*)(ws + 262144);
    float* bufA = (float*)(ws + 262144 + NB);
    float* bufB = (float*)(ws + 262144 + 2 * NB);
    float* osum = (float*)(ws + 262144 + 3 * NB);

    int nnz = in_sizes[2];
    int T = in_sizes[5] / 3;
    int nelem = N_NODES_C * 64;
    int zgrid = (nelem + 255) / 256;
    long sthreads = (long)nnz * 4;
    int sgrid = (int)((sthreads + 255) / 256);

    zero_f32<<<1, 64, 0, stream>>>(scal, 3);
    convert_weights<<<(92160 + 255) / 256, 256, 0, stream>>>(tW, muW, lvW, f1W, f2W, dW,
                                                             tgW, dgW, wb);

    hgcf_dense<<<N_NODES_C / 16, 32, 0, stream>>>(emb_user, emb_item, eps,
                                                  tb, tgb, mub, lvb, f1b, f2b,
                                                  db, dgb, wb,
                                                  mbuf, &scal[0], &scal[1]);

    // 3-layer SpMM with running sum (features stay L2-resident: 25.6 MB << 192 MB)
    zero_f32<<<zgrid, 256, 0, stream>>>(osum, nelem);
    zero_f32<<<zgrid, 256, 0, stream>>>(bufA, nelem);
    spmm_scatter<<<sgrid, 256, 0, stream>>>(adj_rows, adj_cols, adj_vals, mbuf, bufA, nnz);
    accum_add<<<zgrid, 256, 0, stream>>>(osum, bufA, nelem);
    zero_f32<<<zgrid, 256, 0, stream>>>(bufB, nelem);
    spmm_scatter<<<sgrid, 256, 0, stream>>>(adj_rows, adj_cols, adj_vals, bufA, bufB, nnz);
    accum_add<<<zgrid, 256, 0, stream>>>(osum, bufB, nelem);
    zero_f32<<<zgrid, 256, 0, stream>>>(bufA, nelem);
    spmm_scatter<<<sgrid, 256, 0, stream>>>(adj_rows, adj_cols, adj_vals, bufB, bufA, nnz);
    accum_add<<<zgrid, 256, 0, stream>>>(osum, bufA, nelem);

    expmap_rows<<<(N_NODES_C + 255) / 256, 256, 0, stream>>>(osum, bufB, N_NODES_C);
    margin_kernel<<<(T + 255) / 256, 256, 0, stream>>>(triples, bufB, &scal[2], T);
    finalize<<<1, 1, 0, stream>>>(scal, (float*)d_out);
}
